// CustomAttention_67362267070524
// MI455X (gfx1250) — matrix operand order repaired and checked by reading, not compiled, not run on hardware
//
#include <hip/hip_runtime.h>
#include <hip/hip_bf16.h>

// Flash attention, B=2,H=16,S=2048,D=64 fp32, bool keep-mask + causal.
// MI455X: matrix-core bound (~460 flop/byte >> HBM crossover), so:
//   - v_wmma_f32_16x16x32_f16 for QK^T and PV (fp32 accumulate)
//   - double-buffered GLOBAL_LOAD_ASYNC_TO_LDS staging of K/V/mask tiles
//     (ASYNCcnt-tracked) overlapping the next tile's HBM traffic with the
//     current tile's WMMAs + online softmax.

typedef __attribute__((ext_vector_type(16))) _Float16 v16h;
typedef __attribute__((ext_vector_type(8)))  float    v8f;
typedef __attribute__((ext_vector_type(4)))  int      v4i;

#define BM     64
#define BN     64
#define SLEN   2048
#define DHEAD  64
#define HNUM   16
#define KPAD   72            // padded f16 LDS row (16B-aligned rows)
#define NEGBIG (-3.0e38f)

// ---- CDNA5 async global->LDS copy (ASYNCcnt), with portable fallback ------
#if __has_builtin(__builtin_amdgcn_global_load_async_to_lds_b128)
typedef __attribute__((address_space(1))) v4i v4i_g;   // global 16B vector
typedef __attribute__((address_space(3))) v4i v4i_l;   // LDS 16B vector
__device__ __forceinline__ void async_b128(const void* g, void* l) {
    __builtin_amdgcn_global_load_async_to_lds_b128((v4i_g*)g, (v4i_l*)l, 0, 0);
}
__device__ __forceinline__ void async_wait0() {
#if __has_builtin(__builtin_amdgcn_s_wait_asynccnt)
    __builtin_amdgcn_s_wait_asynccnt(0);
#else
    asm volatile("s_wait_asynccnt 0x0" ::: "memory");
#endif
}
#else
__device__ __forceinline__ void async_b128(const void* g, void* l) {
    *(uint4*)l = *(const uint4*)g;   // synchronous fallback through VGPRs
}
__device__ __forceinline__ void async_wait0() {}
#endif

// Issue all async copies for one K/V/mask tile. K/V tiles are contiguous
// 16KB regions (D=64 => rows contiguous): 8 b128 chunks per thread each.
// Mask tile: 64 rows x 64B (row stride SLEN): 2 b128 chunks per thread.
__device__ __forceinline__ void issue_tile(const float* __restrict__ Kh,
                                           const float* __restrict__ Vh,
                                           const unsigned char* __restrict__ Mh,
                                           int i0, int j0, int tid,
                                           float (*Kraw)[DHEAD],
                                           float (*Vraw)[DHEAD],
                                           unsigned char (*Mraw)[BN])
{
    const char* gK = (const char*)(Kh + (size_t)j0 * DHEAD);
    const char* gV = (const char*)(Vh + (size_t)j0 * DHEAD);
    char* lK = (char*)&Kraw[0][0];
    char* lV = (char*)&Vraw[0][0];
    #pragma unroll
    for (int i = 0; i < 8; ++i) {
        const int off = (i * 128 + tid) * 16;
        async_b128(gK + off, lK + off);
        async_b128(gV + off, lV + off);
    }
    #pragma unroll
    for (int i = 0; i < 2; ++i) {
        const int idx = i * 128 + tid;        // 0..255
        const int row = idx >> 2;
        const int c   = (idx & 3) * 16;
        async_b128(Mh + (size_t)(i0 + row) * SLEN + j0 + c, &Mraw[row][c]);
    }
}

__global__ __launch_bounds__(128)
void fa_fwd_kernel(const float* __restrict__ Q, const float* __restrict__ K,
                   const float* __restrict__ V, const unsigned char* __restrict__ Mask,
                   float* __restrict__ O)
{
    __shared__ float Kraw[2][BN][DHEAD];          // async-filled fp32 K tiles
    __shared__ float Vraw[2][BN][DHEAD];          // async-filled fp32 V tiles
    __shared__ unsigned char Mraw[2][BM][BN];     // async-filled mask tiles
    __shared__ _Float16 Ksh[BN][KPAD];            // f16 [key][d]
    __shared__ _Float16 Vsh[DHEAD][KPAD];         // f16 [d][key] (transposed)
    __shared__ _Float16 Psh[4][16][KPAD];         // per-wave P scratch (C->A)

    const int tid      = threadIdx.x;
    const int wave     = tid >> 5;
    const int lane     = tid & 31;
    const int laneRow  = lane & 15;
    const int laneHalf = lane >> 4;
    const int hb       = laneHalf * 8;

    const int i_blk = blockIdx.x;
    const int bh    = blockIdx.y;                 // b*H + h
    const int b     = bh >> 4;                    // H == 16
    const int i0    = i_blk * BM;

    const float scale = 0.125f;                   // 1/sqrt(64)

    const size_t head_off = (size_t)bh * SLEN * DHEAD;
    const float* __restrict__ Qh = Q + head_off;
    const float* __restrict__ Kh = K + head_off;
    const float* __restrict__ Vh = V + head_off;
    const unsigned char* __restrict__ Mh = Mask + (size_t)b * SLEN * SLEN;

    // ---- Q A-fragments (rows i0 + wave*16 + laneRow) ----------------------
    // A (16x32 f16) in this code's contiguous k order, same as B: half 0 holds K {0..15}, half 1 {16..31}.
    v16h qa[2];
    {
        const float* qrow = Qh + (size_t)(i0 + wave * 16 + laneRow) * DHEAD;
        #pragma unroll
        for (int c = 0; c < 2; ++c) {
            #pragma unroll
            for (int i = 0; i < 8; ++i) qa[c][i]     = (_Float16)qrow[c * 32 + laneHalf * 16 + i];
            #pragma unroll
            for (int i = 0; i < 8; ++i) qa[c][8 + i] = (_Float16)qrow[c * 32 + laneHalf * 16 + 8 + i];
        }
    }

    v8f o[4];
    #pragma unroll
    for (int t = 0; t < 4; ++t)
        #pragma unroll
        for (int r = 0; r < 8; ++r) o[t][r] = 0.0f;

    float m_i[8], l_i[8];
    #pragma unroll
    for (int r = 0; r < 8; ++r) { m_i[r] = NEGBIG; l_i[r] = 0.0f; }

    // Prologue: start streaming tile 0 into buffer 0.
    issue_tile(Kh, Vh, Mh, i0, 0, tid, Kraw[0], Vraw[0], Mraw[0]);

    for (int j_blk = 0; j_blk <= i_blk; ++j_blk) {
        const int p  = j_blk & 1;
        const int j0 = j_blk * BN;

        async_wait0();        // our wave's async copies for tile j done
        __syncthreads();      // all waves' copies done; prior-buffer reads fenced

        // Kick off tile j+1 into the other buffer; overlaps with everything below.
        if (j_blk < i_blk)
            issue_tile(Kh, Vh, Mh, i0, j0 + BN, tid,
                       Kraw[1 - p], Vraw[1 - p], Mraw[1 - p]);

        // ---- convert raw fp32 -> f16 K tile and V^T tile ------------------
        {
            const int row  = tid >> 1;
            const int half = tid & 1;
            const float* kr = &Kraw[p][row][half * 32];
            const float* vr = &Vraw[p][row][half * 32];
            #pragma unroll
            for (int i = 0; i < 32; ++i) Ksh[row][half * 32 + i] = (_Float16)kr[i];
            #pragma unroll
            for (int i = 0; i < 32; ++i) Vsh[half * 32 + i][row] = (_Float16)vr[i];
        }
        __syncthreads();

        // ---- S = (Q K^T) * scale, then masks ------------------------------
        v8f s[4];
        #pragma unroll
        for (int t = 0; t < 4; ++t) {
            v8f acc;
            #pragma unroll
            for (int r = 0; r < 8; ++r) acc[r] = 0.0f;
            #pragma unroll
            for (int c = 0; c < 2; ++c) {
                v16h bfrag;   // B 32x16: lane = key column, 16 contiguous K f16
                const _Float16* kr = &Ksh[t * 16 + laneRow][c * 32 + laneHalf * 16];
                #pragma unroll
                for (int i = 0; i < 16; ++i) bfrag[i] = kr[i];
                acc = __builtin_amdgcn_wmma_f32_16x16x32_f16(
                        false, qa[c], false, bfrag, (short)0, acc, false, false);
            }
            s[t] = acc;
        }

        const bool diag = (j_blk == i_blk);
        #pragma unroll
        for (int t = 0; t < 4; ++t) {
            #pragma unroll
            for (int r = 0; r < 8; ++r) {
                const int m_loc = wave * 16 + r + hb;   // query row in tile
                const int n_loc = t * 16 + laneRow;     // key column in tile
                float sv = s[t][r] * scale;
                if (Mraw[p][m_loc][n_loc] == 0) sv = NEGBIG;
                if (diag && (n_loc > m_loc)) sv = NEGBIG;
                s[t][r] = sv;
            }
        }

        // ---- online softmax (butterfly within 16-lane halves) -------------
        #pragma unroll
        for (int r = 0; r < 8; ++r) {
            float rm = fmaxf(fmaxf(s[0][r], s[1][r]), fmaxf(s[2][r], s[3][r]));
            rm = fmaxf(rm, __shfl_xor(rm, 1, 32));
            rm = fmaxf(rm, __shfl_xor(rm, 2, 32));
            rm = fmaxf(rm, __shfl_xor(rm, 4, 32));
            rm = fmaxf(rm, __shfl_xor(rm, 8, 32));
            const float m_new = fmaxf(m_i[r], rm);
            const float alpha = (m_new <= NEGBIG) ? 1.0f : __expf(m_i[r] - m_new);

            float rs = 0.0f;
            #pragma unroll
            for (int t = 0; t < 4; ++t) {
                const float sv = s[t][r];
                const float pv = (sv <= NEGBIG) ? 0.0f : __expf(sv - m_new);
                s[t][r] = pv;
                rs += pv;
            }
            rs += __shfl_xor(rs, 1, 32);
            rs += __shfl_xor(rs, 2, 32);
            rs += __shfl_xor(rs, 4, 32);
            rs += __shfl_xor(rs, 8, 32);

            l_i[r] = l_i[r] * alpha + rs;
            m_i[r] = m_new;
            #pragma unroll
            for (int t = 0; t < 4; ++t) o[t][r] *= alpha;

            #pragma unroll
            for (int t = 0; t < 4; ++t)
                Psh[wave][r + hb][t * 16 + laneRow] = (_Float16)s[t][r];
        }

        // ---- O += P V -----------------------------------------------------
        v16h pa[2];
        {
            const _Float16* pr = &Psh[wave][laneRow][0];
            #pragma unroll
            for (int c = 0; c < 2; ++c) {
                #pragma unroll
                for (int i = 0; i < 8; ++i) pa[c][i]     = pr[c * 32 + laneHalf * 16 + i];
                #pragma unroll
                for (int i = 0; i < 8; ++i) pa[c][8 + i] = pr[c * 32 + laneHalf * 16 + 8 + i];
            }
        }
        #pragma unroll
        for (int t = 0; t < 4; ++t) {
            #pragma unroll
            for (int c = 0; c < 2; ++c) {
                v16h vb;   // B: lane = d column, contiguous keys from V^T
                const _Float16* vr = &Vsh[t * 16 + laneRow][c * 32 + laneHalf * 16];
                #pragma unroll
                for (int i = 0; i < 16; ++i) vb[i] = vr[i];
                o[t] = __builtin_amdgcn_wmma_f32_16x16x32_f16(
                        false, pa[c], false, vb, (short)0, o[t], false, false);
            }
        }
    }

    // ---- normalize and write out ------------------------------------------
    #pragma unroll
    for (int r = 0; r < 8; ++r) {
        const float inv_l = 1.0f / l_i[r];
        const size_t base = head_off + (size_t)(i0 + wave * 16 + r + hb) * DHEAD;
        #pragma unroll
        for (int t = 0; t < 4; ++t)
            O[base + t * 16 + laneRow] = o[t][r] * inv_l;
    }
}

extern "C" void kernel_launch(void* const* d_in, const int* in_sizes, int n_in,
                              void* d_out, int out_size, void* d_ws, size_t ws_size,
                              hipStream_t stream) {
    const float* q = (const float*)d_in[0];
    const float* k = (const float*)d_in[1];
    const float* v = (const float*)d_in[2];
    const unsigned char* mask = (const unsigned char*)d_in[3];  // jnp bool -> 1 byte
    float* out = (float*)d_out;

    dim3 grid(SLEN / BM, 2 * HNUM);   // 32 query tiles x 32 (b,h) heads
    dim3 block(128);                  // 4 wave32 waves
    fa_fwd_kernel<<<grid, block, 0, stream>>>(q, k, v, mask, out);
}
